// KerasPairModel_85289460564401
// MI455X (gfx1250) — matrix operand order repaired and checked by reading, not compiled, not run on hardware
//
#include <hip/hip_runtime.h>

// ---------------------------------------------------------------------------
// MI455X (gfx1250) implementation.
// Compute-bound pair/lig MLPs -> v_wmma_f32_16x16x32_f16 tiles, f32 accum.
// Weights repacked once per call into WMMA B-fragment layout; activations
// staged in LDS between layers; scalar outputs reduced via global atomics.
// ---------------------------------------------------------------------------

typedef __attribute__((ext_vector_type(16))) _Float16 v16h;
typedef __attribute__((ext_vector_type(8)))  _Float16 v8h;
typedef __attribute__((ext_vector_type(8)))  float    v8f;

#define WAVES 4   // 4 waves (128 threads) per block; 16-row tile per wave

// ---- A-fragment loader: 16x32 f16 tile from LDS (row-major, stride Cin) ----
// Layout (ISA 7.12.2): lanes 0-15 -> rows 0-15, K in {0..7,16..23};
//                      lanes 16-31 -> rows 0-15, K in {8..15,24..31}.
__device__ inline v16h load_a_frag(const _Float16* tile, int Cin, int kbase, int lane) {
  const int m   = lane & 15;
  const int off = (lane < 16) ? 0 : 8;
  const _Float16* p = tile + m * Cin + kbase + off;
  v8h lo = *(const v8h*)(p);        // ds_load_b128
  v8h hi = *(const v8h*)(p + 16);   // ds_load_b128
  return __builtin_shufflevector(lo, hi, 0,1,2,3,4,5,6,7,8,9,10,11,12,13,14,15);
}

// ---- one MLP layer: [16 x Kpad] @ [Kpad x Npad], bias + optional ReLU ------
template<int KT, int NT, bool RELU>
__device__ inline void mlp_layer(const _Float16* in, int Cin,
                                 _Float16* out, int Cout,
                                 const _Float16* __restrict__ w,
                                 const float* __restrict__ bias, int lane) {
  const int col = lane & 15;
  const int m0  = (lane < 16) ? 0 : 8;
  #pragma unroll
  for (int n = 0; n < NT; ++n) {
    v8f c = {};
    #pragma unroll
    for (int k = 0; k < KT; ++k) {
      v16h a = load_a_frag(in, Cin, k * 32, lane);
      v16h b = *(const v16h*)(w + (((n * KT + k) << 9) + (lane << 4)));  // 32B/lane
      c = __builtin_amdgcn_wmma_f32_16x16x32_f16(false, a, false, b,
                                                 (short)0, c, false, false);
    }
    const int   nc = n * 16 + col;
    const float bv = bias[nc];
    #pragma unroll
    for (int v = 0; v < 8; ++v) {
      float x = c[v] + bv;
      if (RELU) x = fmaxf(x, 0.0f);
      out[(m0 + v) * Cout + nc] = (_Float16)x;   // C-layout: row = m0+v, col = nc
    }
  }
}

// ---- final layer: Npad=16, only column 0 meaningful; returns C fragment ----
template<int KT>
__device__ inline v8f mlp_last(const _Float16* in, int Cin,
                               const _Float16* __restrict__ w, int lane) {
  v8f c = {};
  #pragma unroll
  for (int k = 0; k < KT; ++k) {
    v16h a = load_a_frag(in, Cin, k * 32, lane);
    v16h b = *(const v16h*)(w + ((k << 9) + (lane << 4)));
    c = __builtin_amdgcn_wmma_f32_16x16x32_f16(false, a, false, b,
                                               (short)0, c, false, false);
  }
  return c;
}

// ---- prep: node features h = [embed[Z](8) | atom_h(32)] as f16 -------------
__global__ void build_h_kernel(const int* __restrict__ Z,
                               const float* __restrict__ atom_h,
                               const float* __restrict__ embed,
                               _Float16* __restrict__ h16, int n) {
  int i = blockIdx.x * blockDim.x + threadIdx.x;
  if (i >= n * 40) return;
  int a = i / 40, j = i - a * 40;
  float v = (j < 8) ? embed[Z[a] * 8 + j] : atom_h[a * 32 + (j - 8)];
  h16[i] = (_Float16)v;
}

// ---- prep: repack row-major fp32 [Ksrc,Nsrc] weight into WMMA B fragments --
// Fragment f = n*ktiles + k holds a 32x16 tile; lane<16 -> K=kbase+{0..7,16..23},
// lane>=16 -> K=kbase+{8..15,24..31}, column = n*16 + (lane&15). Zero-padded.
__global__ void pack_w_kernel(const float* __restrict__ src,
                              _Float16* __restrict__ dst,
                              int Ksrc, int Nsrc, int Kpad, int Npad) {
  const int ktiles = Kpad >> 5;
  const int total  = (Kpad * Npad);
  for (int t = blockIdx.x * blockDim.x + threadIdx.x; t < total;
       t += gridDim.x * blockDim.x) {
    int f = t >> 9, r = t & 511;
    int lane = r >> 4, h = r & 15;
    int n = f / ktiles, k = f - n * ktiles;
    int col = n * 16 + (lane & 15);
    int kk  = k * 32 + ((lane < 16) ? 0 : 8) + ((h < 8) ? h : h + 8);
    float v = (kk < Ksrc && col < Nsrc) ? src[kk * Nsrc + col] : 0.0f;
    dst[t] = (_Float16)v;
  }
}

__global__ void init_kernel(float* accum) { accum[0] = 0.0f; accum[1] = 0.0f; }

__global__ void finish_kernel(const float* __restrict__ accum,
                              const float* __restrict__ shift,
                              float* __restrict__ out) {
  out[0] = accum[0] * 3e-6f - accum[1] * 1e-4f + shift[0];
}

// ---- pair path: per-edge gather + RBF, MLP 96->256->128->64->16(1), sum ----
__global__ __launch_bounds__(WAVES * 32)
void pair_mlp_kernel(const float* __restrict__ RA, const float* __restrict__ RB,
                     const int* __restrict__ e_src, const int* __restrict__ e_tgt,
                     const _Float16* __restrict__ hA16, const _Float16* __restrict__ hB16,
                     const _Float16* __restrict__ w1, const _Float16* __restrict__ w2,
                     const _Float16* __restrict__ w3, const _Float16* __restrict__ w4,
                     const float* __restrict__ b1, const float* __restrict__ b2,
                     const float* __restrict__ b3, const float* __restrict__ b4,
                     float* accum, long E) {
  __shared__ __align__(16) _Float16 sIn[WAVES][16][96];
  __shared__ __align__(16) _Float16 sA [WAVES][16][256];
  __shared__ __align__(16) _Float16 sB [WAVES][16][128];

  const int  lane = threadIdx.x & 31;
  const int  wv   = threadIdx.x >> 5;
  const long tile = (long)blockIdx.x * WAVES + wv;
  const int  r    = lane & 15;
  long e = tile * 16 + r;
  if (e >= E) e = E - 1;               // clamp; invalid rows masked at the end

  const int tgt = e_tgt[e];
  _Float16* row = &sIn[wv][r][0];
  if (lane < 16) {
    const int src = e_src[e];
    const float dx = RB[tgt * 3 + 0] - RA[src * 3 + 0];
    const float dy = RB[tgt * 3 + 1] - RA[src * 3 + 1];
    const float dz = RB[tgt * 3 + 2] - RA[src * 3 + 2];
    const float dR = sqrtf(fmaxf(dx * dx + dy * dy + dz * dz, 0.0f));
    const v8h* hs = (const v8h*)(hA16 + (long)src * 40);
    v8h* hd = (v8h*)row;
    #pragma unroll
    for (int j = 0; j < 5; ++j) hd[j] = hs[j];            // hA -> cols 0..39
    const float fc = (dR < 8.0f)
        ? 0.5f * (__cosf(dR * (3.14159265358979f / 8.0f)) + 1.0f) : 0.0f;
    #pragma unroll
    for (int i = 0; i < 8; ++i) {                          // rbf -> cols 80..87
      float t = dR - (8.0f / 7.0f) * (float)i;
      row[80 + i] = (_Float16)(__expf(-0.5f * t * t) * fc);
    }
    #pragma unroll
    for (int i = 88; i < 96; ++i) row[i] = (_Float16)0.0f; // K padding
  } else {
    const v8h* hs = (const v8h*)(hB16 + (long)tgt * 40);
    v8h* hd = (v8h*)(row + 40);
    #pragma unroll
    for (int j = 0; j < 5; ++j) hd[j] = hs[j];            // hB -> cols 40..79
  }
  __syncthreads();

  mlp_layer<3, 16, true>(&sIn[wv][0][0],  96, &sA[wv][0][0], 256, w1, b1, lane);
  __syncthreads();
  mlp_layer<8,  8, true>(&sA[wv][0][0], 256, &sB[wv][0][0], 128, w2, b2, lane);
  __syncthreads();
  mlp_layer<4,  4, true>(&sB[wv][0][0], 128, &sA[wv][0][0],  64, w3, b3, lane);
  __syncthreads();
  v8f c = mlp_last<2>(&sA[wv][0][0], 64, w4, lane);

  if ((lane & 15) == 0) {              // lanes 0 and 16 hold column 0
    const int   m0 = (lane < 16) ? 0 : 8;
    const float b0 = b4[0];
    float s = 0.0f;
    #pragma unroll
    for (int v = 0; v < 8; ++v) {
      long re = tile * 16 + m0 + v;
      if (re < E) s += c[v] + b0;
    }
    atomicAdd(accum, s);
  }
}

// ---- lig path: per-atom MLP 64->256->128->64->16(1), sum -------------------
__global__ __launch_bounds__(WAVES * 32)
void lig_mlp_kernel(const _Float16* __restrict__ hA16,
                    const _Float16* __restrict__ w1, const _Float16* __restrict__ w2,
                    const _Float16* __restrict__ w3, const _Float16* __restrict__ w4,
                    const float* __restrict__ b1, const float* __restrict__ b2,
                    const float* __restrict__ b3, const float* __restrict__ b4,
                    float* accum, long nA) {
  __shared__ __align__(16) _Float16 sIn[WAVES][16][64];
  __shared__ __align__(16) _Float16 sA [WAVES][16][256];
  __shared__ __align__(16) _Float16 sB [WAVES][16][128];

  const int  lane = threadIdx.x & 31;
  const int  wv   = threadIdx.x >> 5;
  const long tile = (long)blockIdx.x * WAVES + wv;
  const int  r    = lane & 15;
  long a = tile * 16 + r;
  if (a >= nA) a = nA - 1;
  if (lane < 16) {
    _Float16* row = &sIn[wv][r][0];
    const v8h* hs = (const v8h*)(hA16 + a * 40);
    v8h* hd = (v8h*)row;
    #pragma unroll
    for (int j = 0; j < 5; ++j) hd[j] = hs[j];
    #pragma unroll
    for (int i = 40; i < 64; ++i) row[i] = (_Float16)0.0f;  // K padding
  }
  __syncthreads();

  mlp_layer<2, 16, true>(&sIn[wv][0][0],  64, &sA[wv][0][0], 256, w1, b1, lane);
  __syncthreads();
  mlp_layer<8,  8, true>(&sA[wv][0][0], 256, &sB[wv][0][0], 128, w2, b2, lane);
  __syncthreads();
  mlp_layer<4,  4, true>(&sB[wv][0][0], 128, &sA[wv][0][0],  64, w3, b3, lane);
  __syncthreads();
  v8f c = mlp_last<2>(&sA[wv][0][0], 64, w4, lane);

  if ((lane & 15) == 0) {
    const int   m0 = (lane < 16) ? 0 : 8;
    const float b0 = b4[0];
    float s = 0.0f;
    #pragma unroll
    for (int v = 0; v < 8; ++v) {
      long rr = tile * 16 + m0 + v;
      if (rr < nA) s += c[v] + b0;
    }
    atomicAdd(accum, s);
  }
}

// ---------------------------------------------------------------------------
extern "C" void kernel_launch(void* const* d_in, const int* in_sizes, int n_in,
                              void* d_out, int out_size, void* d_ws, size_t ws_size,
                              hipStream_t stream) {
  (void)out_size; (void)ws_size;
  const float* RA      = (const float*)d_in[0];
  const float* RB      = (const float*)d_in[1];
  const float* atom_hA = (const float*)d_in[2];
  const float* atom_hB = (const float*)d_in[3];
  const float* embed   = (const float*)d_in[4];

  const float *pW[4], *pB[4], *lW[4], *lB[4], *shift;
  const int *ZA, *ZB, *e_src, *e_tgt;
  int iZA, iE;
  if (n_in >= 26) {                      // lists flattened to separate inputs
    for (int i = 0; i < 4; ++i) {
      pW[i] = (const float*)d_in[5 + i];
      pB[i] = (const float*)d_in[9 + i];
      lW[i] = (const float*)d_in[13 + i];
      lB[i] = (const float*)d_in[17 + i];
    }
    shift = (const float*)d_in[21];
    ZA = (const int*)d_in[22]; ZB = (const int*)d_in[23];
    e_src = (const int*)d_in[24]; e_tgt = (const int*)d_in[25];
    iZA = 22; iE = 24;
  } else {                               // lists concatenated into one buffer
    const float* pw = (const float*)d_in[5];
    const float* pb = (const float*)d_in[6];
    const float* lw = (const float*)d_in[7];
    const float* lb = (const float*)d_in[8];
    const int pwo[4] = {0, 88*256, 88*256 + 256*128, 88*256 + 256*128 + 128*64};
    const int lwo[4] = {0, 40*256, 40*256 + 256*128, 40*256 + 256*128 + 128*64};
    const int bo[4]  = {0, 256, 384, 448};
    for (int i = 0; i < 4; ++i) {
      pW[i] = pw + pwo[i]; pB[i] = pb + bo[i];
      lW[i] = lw + lwo[i]; lB[i] = lb + bo[i];
    }
    shift = (const float*)d_in[9];
    ZA = (const int*)d_in[10]; ZB = (const int*)d_in[11];
    e_src = (const int*)d_in[12]; e_tgt = (const int*)d_in[13];
    iZA = 10; iE = 12;
  }
  const long nA = in_sizes[iZA];
  const long nB = in_sizes[iZA + 1];
  const long E  = in_sizes[iE];

  // ---- workspace layout (deterministic; rebuilt every call) ----
  char* p = (char*)d_ws;
  auto take = [&](size_t bytes) {
    char* q = p; p += (bytes + 255) & ~(size_t)255; return q;
  };
  _Float16* hA16 = (_Float16*)take((size_t)nA * 40 * 2);
  _Float16* hB16 = (_Float16*)take((size_t)nB * 40 * 2);
  _Float16* pw1  = (_Float16*)take((size_t)96  * 256 * 2);
  _Float16* pw2  = (_Float16*)take((size_t)256 * 128 * 2);
  _Float16* pw3  = (_Float16*)take((size_t)128 * 64  * 2);
  _Float16* pw4  = (_Float16*)take((size_t)64  * 16  * 2);
  _Float16* lw1  = (_Float16*)take((size_t)64  * 256 * 2);
  _Float16* lw2  = (_Float16*)take((size_t)256 * 128 * 2);
  _Float16* lw3  = (_Float16*)take((size_t)128 * 64  * 2);
  _Float16* lw4  = (_Float16*)take((size_t)64  * 16  * 2);
  float*    accum = (float*)take(2 * sizeof(float));

  init_kernel<<<1, 1, 0, stream>>>(accum);

  build_h_kernel<<<(int)((nA * 40 + 255) / 256), 256, 0, stream>>>(ZA, atom_hA, embed, hA16, (int)nA);
  build_h_kernel<<<(int)((nB * 40 + 255) / 256), 256, 0, stream>>>(ZB, atom_hB, embed, hB16, (int)nB);

  auto packgrid = [](int kp, int np_) { return (kp * np_ + 255) / 256; };
  pack_w_kernel<<<packgrid(96, 256), 256, 0, stream>>>(pW[0], pw1,  88, 256,  96, 256);
  pack_w_kernel<<<packgrid(256,128), 256, 0, stream>>>(pW[1], pw2, 256, 128, 256, 128);
  pack_w_kernel<<<packgrid(128, 64), 256, 0, stream>>>(pW[2], pw3, 128,  64, 128,  64);
  pack_w_kernel<<<packgrid(64,  16), 256, 0, stream>>>(pW[3], pw4,  64,   1,  64,  16);
  pack_w_kernel<<<packgrid(64, 256), 256, 0, stream>>>(lW[0], lw1,  40, 256,  64, 256);
  pack_w_kernel<<<packgrid(256,128), 256, 0, stream>>>(lW[1], lw2, 256, 128, 256, 128);
  pack_w_kernel<<<packgrid(128, 64), 256, 0, stream>>>(lW[2], lw3, 128,  64, 128,  64);
  pack_w_kernel<<<packgrid(64,  16), 256, 0, stream>>>(lW[3], lw4,  64,   1,  64,  16);

  const int ptiles = (int)((E + 15) / 16);
  pair_mlp_kernel<<<(ptiles + WAVES - 1) / WAVES, WAVES * 32, 0, stream>>>(
      RA, RB, e_src, e_tgt, hA16, hB16,
      pw1, pw2, pw3, pw4, pB[0], pB[1], pB[2], pB[3], accum + 0, E);

  const int ltiles = (int)((nA + 15) / 16);
  lig_mlp_kernel<<<(ltiles + WAVES - 1) / WAVES, WAVES * 32, 0, stream>>>(
      hA16, lw1, lw2, lw3, lw4, lB[0], lB[1], lB[2], lB[3], accum + 1, nA);

  finish_kernel<<<1, 1, 0, stream>>>(accum, shift, (float*)d_out);
}
